// Diff_Encoder_33732673143025
// MI455X (gfx1250) — matrix operand order repaired and checked by reading, not compiled, not run on hardware
//
#include <hip/hip_runtime.h>
#include <hip/hip_bf16.h>

#define NN     100000
#define DIN    256
#define HID    128
#define NEDGE  1600000
#define EPSV   1e-5f

typedef __attribute__((ext_vector_type(16))) __bf16 v16bf;
typedef __attribute__((ext_vector_type(8)))  __bf16 v8bf;
typedef __attribute__((ext_vector_type(8)))  float  v8f;
typedef __attribute__((ext_vector_type(4)))  int    v4i;

union V16U { v16bf v; v8bf h[2]; };

#if defined(__has_builtin)
#if __has_builtin(__builtin_amdgcn_global_load_async_to_lds_b128)
#define HAVE_ASYNC_LDS 1
#endif
#endif

__device__ __forceinline__ unsigned short f2bf(float f) {
    unsigned int u = __float_as_uint(f);
    unsigned int r = u + 0x7FFFu + ((u >> 16) & 1u);   // round-to-nearest-even
    return (unsigned short)(r >> 16);
}

// ---------------- f32 -> bf16 conversion (row-major, for A matrices) ----------------
__global__ void k_cvt_bf16(const float* __restrict__ in,
                           unsigned short* __restrict__ out, long n) {
    long i = (long)blockIdx.x * blockDim.x + threadIdx.x;
    if (i < n) out[i] = f2bf(in[i]);
}

// ---------------- pack W (K x HID f32) into WMMA B-fragment order (bf16) ----------------
// packed[(((ntile*(K/32) + kb)*32 + lane)*16 + j] = bf16( W[kb*32 + (lane<16?0:16) + j][ntile*16 + (lane&15)] )
// => each lane's 16-element fragment is 32 contiguous bytes.
__global__ void k_pack_b(const float* __restrict__ W, unsigned short* __restrict__ out, int K) {
    int idx = blockIdx.x * blockDim.x + threadIdx.x;
    if (idx >= K * HID) return;
    const int j    = idx & 15;
    const int lane = (idx >> 4) & 31;
    const int rest = idx >> 9;
    const int kblocks = K >> 5;
    const int kb    = rest % kblocks;
    const int ntile = rest / kblocks;
    const int k = kb * 32 + ((lane < 16) ? 0 : 16) + j;
    const int n = ntile * 16 + (lane & 15);
    out[idx] = f2bf(W[(long)k * HID + n]);
}

// ---------------- zero fill ----------------
__global__ void k_zero(float* __restrict__ p, long n) {
    long i = (long)blockIdx.x * blockDim.x + threadIdx.x;
    if (i < n) p[i] = 0.0f;
}

// ---------------- WMMA bf16 GEMM: C[M x HID] = A[M x K] * Bpacked ----------------
// Block: 256 threads = 8 waves. Block stages the whole packed B (<=64KB) in LDS
// via gfx1250 async-to-LDS DMA (ASYNCcnt) when available.
// Each wave computes one 16-row m-tile across all 8 n-tiles (16x128 of C):
// per k-step: 2 global b128 loads (A) + 8x (2 LDS b128 reads + WMMA).
__global__ __launch_bounds__(256) void k_gemm_bf16(
    const unsigned short* __restrict__ A,
    const unsigned short* __restrict__ Bp,
    float* __restrict__ C, int M, int K)
{
    __shared__ unsigned short lds[DIN * HID];   // 64KB max (K=256)

    const int bytes = K * HID * 2;
#if HAVE_ASYNC_LDS
    {
        typedef __attribute__((address_space(1))) v4i g_v4i;
        typedef __attribute__((address_space(3))) v4i l_v4i;
        char* gbase = (char*)const_cast<unsigned short*>(Bp);
        char* lbase = (char*)lds;
        for (int i = threadIdx.x * 16; i < bytes; i += 256 * 16)
            __builtin_amdgcn_global_load_async_to_lds_b128(
                (g_v4i*)(gbase + i), (l_v4i*)(lbase + i), 0, 0);
#if __has_builtin(__builtin_amdgcn_s_wait_asynccnt)
        __builtin_amdgcn_s_wait_asynccnt(0);
#else
        asm volatile("s_wait_asynccnt 0x0" ::: "memory");
#endif
    }
#else
    for (int i = threadIdx.x * 8; i < bytes / 2; i += 256 * 8)
        *(uint4*)(lds + i) = *(const uint4*)(Bp + i);
#endif
    __syncthreads();

    const int lane  = threadIdx.x & 31;
    const int wave  = threadIdx.x >> 5;
    const int mtile = blockIdx.x * 8 + wave;
    if (mtile >= (M >> 4)) return;          // wave-uniform; EXEC stays all-1s

    const int m       = lane & 15;
    const int khalf   = (lane < 16) ? 0 : 16;
    const int n       = lane & 15;
    const int kblocks = K >> 5;

    const unsigned short* arow = A + (long)(mtile * 16 + m) * K;

    v8f acc[8];
    #pragma unroll
    for (int nt = 0; nt < 8; ++nt) acc[nt] = (v8f){};

    for (int kb = 0; kb < kblocks; ++kb) {
        const int kk = kb * 32;
        if (kk + 32 < K) __builtin_prefetch(arow + kk + 32, 0, 3);  // near-scope prefetch
        // A fragment: two aligned 16B chunks of consecutive K
        V16U a;
        a.h[0] = *(const v8bf*)(arow + kk + khalf);
        a.h[1] = *(const v8bf*)(arow + kk + khalf + 8);
        #pragma unroll
        for (int nt = 0; nt < 8; ++nt) {
            const v16bf b = *(const v16bf*)(lds + ((nt * kblocks + kb) * 32 + lane) * 16);
            acc[nt] = __builtin_amdgcn_wmma_f32_16x16x32_bf16(
                false, a.v, false, b, (short)0, acc[nt], false, false);
        }
    }

    // C/D layout: VGPR v -> row M = v (lanes 0-15) or v+8 (lanes 16-31)
    const int mrow = mtile * 16 + ((lane < 16) ? 0 : 8);
    float* cbase = C + (long)mrow * HID + n;
    #pragma unroll
    for (int nt = 0; nt < 8; ++nt) {
        float* crow = cbase + nt * 16;
        #pragma unroll
        for (int v = 0; v < 8; ++v)
            crow[(long)v * HID] = acc[nt][v];
    }
}

// ---------------- edge scatter: agg[dst] += w_e * h[src] ----------------
// One wave per edge; each lane handles 4 consecutive floats (32*4 = 128).
__global__ __launch_bounds__(256) void k_scatter(
    const long long* __restrict__ src, const long long* __restrict__ dst,
    const float* __restrict__ w, const float* __restrict__ h,
    float* __restrict__ agg, int nedges)
{
    int e = blockIdx.x * 8 + (threadIdx.x >> 5);
    if (e >= nedges) return;
    const int lane = threadIdx.x & 31;
    const long s = (long)src[e];
    const long d = (long)dst[e];
    const float we = w[e];
    const float4 v = *((const float4*)(h + s * HID) + lane);
    float* arow = agg + d * HID + (long)lane * 4;
    atomicAdd(arow + 0, we * v.x);
    atomicAdd(arow + 1, we * v.y);
    atomicAdd(arow + 2, we * v.z);
    atomicAdd(arow + 3, we * v.w);
}

// ---------------- per-column sum / sum-of-squares ----------------
__global__ __launch_bounds__(128) void k_stats(
    const float* __restrict__ h, float* __restrict__ stats, int rows)
{
    const int col = threadIdx.x;
    const int rpb = (rows + gridDim.x - 1) / gridDim.x;
    const int r0 = blockIdx.x * rpb;
    int r1 = r0 + rpb; if (r1 > rows) r1 = rows;
    float s = 0.0f, sq = 0.0f;
    for (int r = r0; r < r1; ++r) {
        float v = h[(long)r * HID + col];
        s += v; sq += v * v;
    }
    atomicAdd(&stats[col], s);
    atomicAdd(&stats[HID + col], sq);
}

// ---------------- BN (batch stats) + ReLU; optional f32 and/or bf16 output ----------------
__global__ void k_bn_relu(const float* __restrict__ h, const float* __restrict__ stats,
                          const float* __restrict__ gamma, const float* __restrict__ beta,
                          float* __restrict__ out_f32, unsigned short* __restrict__ out_bf16,
                          long total)
{
    long i = (long)blockIdx.x * blockDim.x + threadIdx.x;
    if (i >= total) return;
    const int col = (int)(i & (HID - 1));
    const float inv_n = 1.0f / (float)NN;
    float mean = stats[col] * inv_n;
    float var  = stats[HID + col] * inv_n - mean * mean;   // biased variance
    float v = (h[i] - mean) * rsqrtf(var + EPSV) * gamma[col] + beta[col];
    v = fmaxf(v, 0.0f);
    if (out_f32)  out_f32[i]  = v;
    if (out_bf16) out_bf16[i] = f2bf(v);
}

extern "C" void kernel_launch(void* const* d_in, const int* in_sizes, int n_in,
                              void* d_out, int out_size, void* d_ws, size_t ws_size,
                              hipStream_t stream) {
    const float*     x   = (const float*)d_in[0];
    const long long* ei  = (const long long*)d_in[1];   // int64, (2,E) flat
    const float*     ew  = (const float*)d_in[2];
    const float*     W1  = (const float*)d_in[3];
    // d_in[4] = b1, d_in[8] = b2: constant per-column shifts cancel in BatchNorm
    const float*     g1  = (const float*)d_in[5];
    const float*     be1 = (const float*)d_in[6];
    const float*     W2  = (const float*)d_in[7];
    const float*     g2  = (const float*)d_in[9];
    const float*     be2 = (const float*)d_in[10];
    float* out = (float*)d_out;

    const long long* e_src = ei;
    const long long* e_dst = ei + NEDGE;

    // ---- workspace layout (256B aligned) ----
    char* ws = (char*)d_ws;
    size_t off = 0;
    auto alloc = [&](size_t bytes) { char* p = ws + off; off = (off + bytes + 255) & ~(size_t)255; return p; };
    unsigned short* xb  = (unsigned short*)alloc((size_t)NN * DIN * 2);   // x in bf16 (row-major)
    unsigned short* hb  = (unsigned short*)alloc((size_t)NN * HID * 2);   // BN+ReLU out (bf16)
    unsigned short* w1p = (unsigned short*)alloc((size_t)DIN * HID * 2);  // W1 packed fragments
    unsigned short* w2p = (unsigned short*)alloc((size_t)HID * HID * 2);  // W2 packed fragments
    float*          hf  = (float*)alloc((size_t)NN * HID * 4);            // GEMM output
    float*          ag  = (float*)alloc((size_t)NN * HID * 4);            // scatter accumulator
    float*          st  = (float*)alloc(2 * HID * 4);                     // sums / sumsq
    (void)ws_size;

    const long nx = (long)NN * DIN;   // 25.6M
    const long nh = (long)NN * HID;   // 12.8M
    const int  gemm_blocks = (NN / 16 + 7) / 8;              // 782 (6250 m-tiles)
    const int  scat_blocks = (NEDGE + 7) / 8;                // 200000

    // precision conversion + fragment packing
    k_cvt_bf16<<<(unsigned)((nx + 255) / 256), 256, 0, stream>>>(x, xb, nx);
    k_pack_b<<<(DIN * HID + 255) / 256, 256, 0, stream>>>(W1, w1p, DIN);
    k_pack_b<<<(HID * HID + 255) / 256, 256, 0, stream>>>(W2, w2p, HID);

    // ---------- Layer 1 ----------
    k_gemm_bf16<<<gemm_blocks, 256, 0, stream>>>(xb, w1p, hf, NN, DIN);
    k_zero<<<(unsigned)((nh + 255) / 256), 256, 0, stream>>>(ag, nh);
    k_zero<<<1, 256, 0, stream>>>(st, 2 * HID);
    k_scatter<<<scat_blocks, 256, 0, stream>>>(e_src, e_dst, ew, hf, ag, NEDGE);
    k_stats<<<1024, 128, 0, stream>>>(ag, st, NN);
    k_bn_relu<<<(unsigned)((nh + 255) / 256), 256, 0, stream>>>(ag, st, g1, be1, nullptr, hb, nh);

    // ---------- Layer 2 ----------
    k_gemm_bf16<<<gemm_blocks, 256, 0, stream>>>(hb, w2p, hf, NN, HID);
    k_zero<<<(unsigned)((nh + 255) / 256), 256, 0, stream>>>(ag, nh);
    k_zero<<<1, 256, 0, stream>>>(st, 2 * HID);
    k_scatter<<<scat_blocks, 256, 0, stream>>>(e_src, e_dst, ew, hf, ag, NEDGE);
    k_stats<<<1024, 128, 0, stream>>>(ag, st, NN);
    k_bn_relu<<<(unsigned)((nh + 255) / 256), 256, 0, stream>>>(ag, st, g2, be2, out, nullptr, nh);
}
